// YOSOAttention_63926293233878
// MI455X (gfx1250) — matrix operand order repaired and checked by reading, not compiled, not run on hardware
//
#include <hip/hip_runtime.h>
#include <hip/hip_bf16.h>
#include <math.h>

// Problem constants (from reference setup_inputs)
#define BATCH 2
#define HEADS 12
#define SEQ   2048
#define DIM   64
#define BH    (BATCH * HEADS)
#define NROWS (BH * SEQ)          // 49152
#define EPSN 1e-12f

typedef __attribute__((ext_vector_type(16))) _Float16 v16h;
typedef __attribute__((ext_vector_type(8)))  _Float16 v8h;
typedef __attribute__((ext_vector_type(8)))  float    v8f;

// ---------------------------------------------------------------------------
// Kernel 1: l2-normalize Q and K rows -> f16. One wave (32 lanes) per row of 64.
// ---------------------------------------------------------------------------
__global__ __launch_bounds__(256) void prep_qk_kernel(
    const float* __restrict__ Q, const float* __restrict__ K,
    _Float16* __restrict__ qh, _Float16* __restrict__ kh)
{
    const int row  = blockIdx.x * 8 + (threadIdx.x >> 5);
    const int lane = threadIdx.x & 31;
    if (row >= NROWS) return;

    // ---- Q ----
    {
        const float2 q = ((const float2*)(Q + (size_t)row * DIM))[lane];
        float ss = q.x * q.x + q.y * q.y;
        #pragma unroll
        for (int off = 16; off >= 1; off >>= 1) ss += __shfl_xor(ss, off, 32);
        const float inv =
            __builtin_amdgcn_rcpf(fmaxf(__builtin_amdgcn_sqrtf(ss), EPSN));
        union { _Float16 h[2]; unsigned int u; } pk;
        pk.h[0] = (_Float16)(q.x * inv);
        pk.h[1] = (_Float16)(q.y * inv);
        ((unsigned int*)qh)[(size_t)row * 32 + lane] = pk.u;
    }
    // ---- K ----
    {
        const float2 k = ((const float2*)(K + (size_t)row * DIM))[lane];
        float ss = k.x * k.x + k.y * k.y;
        #pragma unroll
        for (int off = 16; off >= 1; off >>= 1) ss += __shfl_xor(ss, off, 32);
        const float inv =
            __builtin_amdgcn_rcpf(fmaxf(__builtin_amdgcn_sqrtf(ss), EPSN));
        union { _Float16 h[2]; unsigned int u; } pk;
        pk.h[0] = (_Float16)(k.x * inv);
        pk.h[1] = (_Float16)(k.y * inv);
        ((unsigned int*)kh)[(size_t)row * 32 + lane] = pk.u;
    }
}

// ---------------------------------------------------------------------------
// Kernel 2: vt[bh][d][t] = f16( V[bh][t][d] * mask[b][t] )   (64x64 LDS tiles)
// ---------------------------------------------------------------------------
__global__ __launch_bounds__(256) void prep_vt_kernel(
    const float* __restrict__ V, const float* __restrict__ mask,
    _Float16* __restrict__ vt)
{
    __shared__ _Float16 tile[DIM][DIM + 1];
    const int bh = blockIdx.y;
    const int b  = bh / HEADS;
    const int t0 = blockIdx.x * 64;
    const int c  = threadIdx.x & 63;       // d index on load, t index on store
    const int rg = threadIdx.x >> 6;       // 0..3

    #pragma unroll
    for (int rr = 0; rr < 16; ++rr) {
        const int r = rg * 16 + rr;        // local t
        const float mv = mask[(size_t)b * SEQ + t0 + r];
        const float v  = V[((size_t)bh * SEQ + t0 + r) * DIM + c] * mv;
        tile[c][r] = (_Float16)v;
    }
    __syncthreads();
    #pragma unroll
    for (int dd = 0; dd < 16; ++dd) {
        const int d = rg * 16 + dd;
        vt[((size_t)bh * DIM + d) * SEQ + t0 + c] = tile[d][c];
    }
}

// ---------------------------------------------------------------------------
// Kernel 3: fused YOSO attention (no LDS, no cross-wave sync).
//   Block = 128 threads (4 waves); block covers 64 query rows of one (b,h);
//   each wave owns 16 query rows with the 16x64 X stripe in 4 WMMA accumulators.
//
//   Per 32-key iteration:
//     S^T = Kn(32x64) . Qn^T            -> 4x v_wmma_f32_16x16x32_f16
//           (A = K fragments, B = persistent Q fragments)
//     W   = (1 - acos(S)/pi)^8          single-sqrt branch-free approx
//     The S^T C-tile layout IS the W A-fragment layout for GEMM2:
//           lane(g,l16): subtile st, vgpr j  ->  W[m=l16, k=16*st+8g+j]
//     X  += W(16x32) . Vm(32x64)        -> 4x v_wmma_f32_16x16x32_f16
//   Epilogue: query mask, row l2-norm, + depthwise conv3 over sequence.
// ---------------------------------------------------------------------------
__global__ __launch_bounds__(128) void yoso_attn_kernel(
    const _Float16* __restrict__ qh, const _Float16* __restrict__ kh,
    const _Float16* __restrict__ vt, const float* __restrict__ V,
    const float* __restrict__ mask, const float* __restrict__ convw,
    float* __restrict__ out)
{
    const int bh   = blockIdx.y;
    const int b    = bh / HEADS;
    const int h    = bh % HEADS;
    const int wave = threadIdx.x >> 5;
    const int lane = threadIdx.x & 31;
    const int g    = lane >> 4;             // lane half-group: 0 or 1
    const int l16  = lane & 15;

    const int m0 = blockIdx.x * 64 + wave * 16;   // first query row of this wave

    const _Float16* __restrict__ qb = qh + (size_t)bh * SEQ * DIM;
    const _Float16* __restrict__ kb = kh + (size_t)bh * SEQ * DIM;
    const _Float16* __restrict__ vb = vt + (size_t)bh * DIM * SEQ;

    // --- persistent B fragments of Qn^T (32 d x 16 queries) for 2 D halves ---
    // B-layout (16-bit 32x16): lane: col n = lane%16 (query m0+l16);
    // halves = K(d-local) = 16g..16g+15 contiguous -> one 32B load.
    v16h bq[2];
    #pragma unroll
    for (int c = 0; c < 2; ++c) {
        bq[c] = *(const v16h*)(qb + (size_t)(m0 + l16) * DIM + 32 * c + 16 * g);
    }

    v8f xacc[4] = { {}, {}, {}, {} };       // X stripe: 4 col tiles of 16

    // acos(d)/pi ~= sqrt(1-d) * (G0 + G1 d + G2 d^2 + G3 d^3)  over d in [-1,1].
    // Fit interpolates d = {1, 0.5, 0, -0.5}: exact where W=u^8 is significant;
    // the d ~ -1 region (where the fit is loose) has W < 1e-6 -> irrelevant.
    const float G0 =  0.5000000f;
    const float G1 = -0.0701310f;
    const float G2 =  0.0314680f;
    const float G3 = -0.0111790f;

    for (int t0 = 0; t0 < SEQ; t0 += 32) {
        // prefetch next key/value tiles (speculative; dropped if OOB)
        __builtin_prefetch(kb + (size_t)(t0 + 32) * DIM, 0, 1);
        __builtin_prefetch(vb + (size_t)(t0 + 32), 0, 1);

        // ---- GEMM1: S^T[32keys x 16queries] = Kn . Qn^T --------------------
        // A-layout (16-bit 16x32): lane: row = key t0+16*st+l16;
        // halves0..15 = d 32c+16g..32c+16g+15 contiguous
        v8f s[2] = { {}, {} };
        #pragma unroll
        for (int st = 0; st < 2; ++st) {
            #pragma unroll
            for (int c = 0; c < 2; ++c) {
                const _Float16* p =
                    kb + (size_t)(t0 + 16 * st + l16) * DIM + 32 * c + 16 * g;
                const v16h ak = *(const v16h*)p;
                s[st] = __builtin_amdgcn_wmma_f32_16x16x32_f16(
                    false, ak, false, bq[c], (short)0, s[st], false, false);
            }
        }

        // ---- elementwise W = (1 - acos(d)/pi)^8, packed straight into the --
        // ---- A fragment for GEMM2 (no LDS, no branches, one sqrt) ----------
        v16h aw;
        #pragma unroll
        for (int st = 0; st < 2; ++st) {
            #pragma unroll
            for (int j = 0; j < 8; ++j) {
                const float d = s[st][j];
                const float r =
                    __builtin_amdgcn_sqrtf(fmaxf(1.0f - d, 0.0f));
                float p = fmaf(G3, d, G2);
                p = fmaf(p, d, G1);
                p = fmaf(p, d, G0);
                const float u = fmaf(-r, p, 1.0f);     // 1 - acos(d)/pi
                const float u2 = u * u;
                const float u4 = u2 * u2;
                aw[st * 8 + j] = (_Float16)(u4 * u4);  // u^8
            }
        }

        // ---- GEMM2: X[16 x 64] += W . Vm  (vt is pre-transposed) -----------
        #pragma unroll
        for (int ct = 0; ct < 4; ++ct) {
            const _Float16* p =
                vb + (size_t)(16 * ct + l16) * SEQ + t0 + 8 * g;
            const v8h blo = *(const v8h*)p;
            const v8h bhi = *(const v8h*)(p + 16);
            const v16h bv = __builtin_shufflevector(   // col n = d, halves = keys
                blo, bhi, 0,1,2,3,4,5,6,7,8,9,10,11,12,13,14,15);
            xacc[ct] = __builtin_amdgcn_wmma_f32_16x16x32_f16(
                false, aw, false, bv, (short)0, xacc[ct], false, false);
        }
    }

    // ---- epilogue: query mask, row l2norm, depthwise conv3, store ----------
    const float w0 = convw[h * 3 + 0];
    const float w1 = convw[h * 3 + 1];
    const float w2 = convw[h * 3 + 2];
    const float* __restrict__ mrow = mask + (size_t)b * SEQ;

    #pragma unroll
    for (int j = 0; j < 8; ++j) {
        const int m = m0 + j + 8 * g;               // global query row
        const float mq = mrow[m];

        // row sum of (x*mq)^2 : partial over this lane's 4 col tiles
        float p = 0.0f;
        #pragma unroll
        for (int ct = 0; ct < 4; ++ct) {
            const float v = xacc[ct][j] * mq;
            p += v * v;
        }
        // reduce across the 16 lanes sharing this row (xor<16 stays in group)
        #pragma unroll
        for (int off = 8; off >= 1; off >>= 1) p += __shfl_xor(p, off, 32);
        const float inv =
            __builtin_amdgcn_rcpf(fmaxf(__builtin_amdgcn_sqrtf(p), EPSN));

        // conv inputs (Vm = V * mask over sequence axis)
        const float mmL = (m > 0)       ? mrow[m - 1] : 0.0f;
        const float mmR = (m < SEQ - 1) ? mrow[m + 1] : 0.0f;

        #pragma unroll
        for (int ct = 0; ct < 4; ++ct) {
            const int n = 16 * ct + l16;
            const size_t base = ((size_t)bh * SEQ + m) * DIM + n;
            float cv = w1 * V[base] * mq;
            if (m > 0)       cv += w0 * V[base - DIM] * mmL;
            if (m < SEQ - 1) cv += w2 * V[base + DIM] * mmR;
            out[base] = xacc[ct][j] * mq * inv + cv;
        }
    }
}

// ---------------------------------------------------------------------------
extern "C" void kernel_launch(void* const* d_in, const int* in_sizes, int n_in,
                              void* d_out, int out_size, void* d_ws, size_t ws_size,
                              hipStream_t stream)
{
    const float* Q     = (const float*)d_in[0];
    const float* K     = (const float*)d_in[1];
    const float* V     = (const float*)d_in[2];
    const float* mask  = (const float*)d_in[3];
    const float* convw = (const float*)d_in[4];
    float* out = (float*)d_out;

    const size_t elems = (size_t)NROWS * DIM;   // 3,145,728
    _Float16* qh = (_Float16*)d_ws;
    _Float16* kh = qh + elems;
    _Float16* vt = kh + elems;

    prep_qk_kernel<<<NROWS / 8, 256, 0, stream>>>(Q, K, qh, kh);
    prep_vt_kernel<<<dim3(SEQ / 64, BH), 256, 0, stream>>>(V, mask, vt);
    yoso_attn_kernel<<<dim3(SEQ / 64, BH), 128, 0, stream>>>(
        qh, kh, vt, V, mask, convw, out);
}
